// DECSeq2_41180146434807
// MI455X (gfx1250) — compile-verified, matrix operand order checked by reading, not run on hardware
//
#include <hip/hip_runtime.h>

// ---------------------------------------------------------------------------
// Problem constants (from reference)
// ---------------------------------------------------------------------------
#define BB   512      // batches
#define LL   128      // sequence length
#define MM   127      // edges per sequence (L-1)
#define MP   128      // padded M for 16-tiling
#define KNN  5        // neighbors
#define EPS  1e-5f

typedef __attribute__((ext_vector_type(16))) _Float16 v16h;
typedef __attribute__((ext_vector_type(8)))  float    v8f;

// ---------------------------------------------------------------------------
// WMMA fragment helpers (wave32, v_wmma_f32_16x16x32_f16)
// A fragment: 16x32 f16, row-major source with given stride (in halves).
//   lane<16 : row=lane,    elems 0..7 = K 0..7,  elems 8..15 = K 16..23
//   lane>=16: row=lane-16, elems 0..7 = K 8..15, elems 8..15 = K 24..31
// B fragments are loaded with the SAME pattern from weights that were
// pre-packed tile-column-major (N x K inside each 32x16 tile), so every
// fragment load is two contiguous 16-byte chunks per lane.
// ---------------------------------------------------------------------------
__device__ inline v16h load_frag_a(const _Float16* base, int stride) {
  int lane = threadIdx.x & 31;
  int row  = lane & 15;
  int kb   = (lane >> 4) * 8;
  const _Float16* p = base + row * stride + kb;
  v16h f;
#pragma unroll
  for (int j = 0; j < 8; ++j) { f[j] = p[j]; f[8 + j] = p[16 + j]; }
  return f;
}

__device__ inline v8f wmma_f16(v16h a, v16h b, v8f c) {
  return __builtin_amdgcn_wmma_f32_16x16x32_f16(false, a, false, b,
                                                (short)0, c, false, false);
}

// ---------------------------------------------------------------------------
// K0: f32 KxN weights -> f16 fragment-ready packed tiles.
// Tile (kt, ct) covers K rows [kt*32, kt*32+32) and cols [ct*16, ct*16+16).
// dst[((kt*tn + ct)*16 + col)*32 + kk] = src[k*N + n]
// At use: b_frag = load_frag_a(dst + tile*512, 32)
// ---------------------------------------------------------------------------
__global__ void pack_b_kernel(const float* __restrict__ src,
                              _Float16* __restrict__ dst, int Kd, int N) {
  int gid = blockIdx.x * 256 + threadIdx.x;
  if (gid >= Kd * N) return;
  int k = gid / N, n = gid % N;
  int kt = k >> 5, kk = k & 31;
  int ct = n >> 4, col = n & 15;
  int tn = N >> 4;
  dst[(size_t)((kt * tn + ct) * 16 + col) * 32 + kk] = (_Float16)src[gid];
}

// ---------------------------------------------------------------------------
// K1: fused bidirectional conv (6->64) + BN1 + ReLU + directional max.
// Output index (b,m) maps to physical batch c=B-1-b, edge e0=M-1-m.
//   branch1: sum_t W[t] . F[c, e0 + 2 - t]
//   branch2: sum_t W[t] . F[c, e0 + t - 2]
// F[c,e] = [ pos[c*L+e+1]-pos[c*L+e] , pos[c*L+e] ]  (6 dims, zero-padded)
// ---------------------------------------------------------------------------
__global__ __launch_bounds__(256) void conv_fuse_kernel(
    const float* __restrict__ pos, const float* __restrict__ w,
    const float* __restrict__ bias, const float* __restrict__ g1,
    const float* __restrict__ b1, const float* __restrict__ m1,
    const float* __restrict__ v1, float* __restrict__ x1,
    _Float16* __restrict__ xf16, _Float16* __restrict__ h16) {
  int gid = blockIdx.x * 256 + threadIdx.x;
  if (gid >= BB * MP * 64) return;
  int ch = gid & 63;
  int pid = gid >> 6;
  int b = pid >> 7;
  int m = pid & 127;
  if (m == MM) {  // padded row -> zeros
    x1[pid * 64 + ch] = 0.f;
    xf16[pid * 64 + ch] = (_Float16)0.f;
    h16[pid * 192 + ch] = (_Float16)0.f;
    return;
  }
  int c  = BB - 1 - b;
  int e0 = (MM - 1) - m;
  float s1 = 0.f, s2 = 0.f;
#pragma unroll
  for (int t = 0; t < 5; ++t) {
    int ea = e0 + 2 - t;
    int eb = e0 + t - 2;
    float Fa[6], Fb[6];
#pragma unroll
    for (int d = 0; d < 6; ++d) { Fa[d] = 0.f; Fb[d] = 0.f; }
    if (ea >= 0 && ea < MM) {
      int base = (c * LL + ea) * 3;
#pragma unroll
      for (int d = 0; d < 3; ++d) {
        float p0 = pos[base + d];
        float p1 = pos[base + 3 + d];
        Fa[d] = p1 - p0;
        Fa[3 + d] = p0;
      }
    }
    if (eb >= 0 && eb < MM) {
      int base = (c * LL + eb) * 3;
#pragma unroll
      for (int d = 0; d < 3; ++d) {
        float p0 = pos[base + d];
        float p1 = pos[base + 3 + d];
        Fb[d] = p1 - p0;
        Fb[3 + d] = p0;
      }
    }
#pragma unroll
    for (int i = 0; i < 6; ++i) {
      float wv = w[(t * 6 + i) * 64 + ch];
      s1 = fmaf(wv, Fa[i], s1);
      s2 = fmaf(wv, Fb[i], s2);
    }
  }
  float sc = rsqrtf(v1[ch] + EPS) * g1[ch];
  float r1 = fmaxf((s1 + bias[ch] - m1[ch]) * sc + b1[ch], 0.f);
  float r2 = fmaxf((s2 + bias[ch] - m1[ch]) * sc + b1[ch], 0.f);
  float out = fmaxf(r1, r2);
  x1[pid * 64 + ch] = out;
  xf16[pid * 64 + ch] = (_Float16)out;
  h16[pid * 192 + ch] = (_Float16)out;
}

// ---------------------------------------------------------------------------
// K1b: per-point squared norms
// ---------------------------------------------------------------------------
__global__ void sq_kernel(const float* __restrict__ x1, float* __restrict__ sq) {
  int pid = blockIdx.x * 256 + threadIdx.x;
  if (pid >= BB * MP) return;
  float s = 0.f;
  const float* p = x1 + pid * 64;
#pragma unroll
  for (int c = 0; c < 64; ++c) s = fmaf(p[c], p[c], s);
  sq[pid] = s;
}

// ---------------------------------------------------------------------------
// K2: per-batch Gram matrix via WMMA (chunks of 32 rows) + stable top-5.
// G = X X^T so the B tile is just another row-slab of X -> same frag load.
// ---------------------------------------------------------------------------
__global__ __launch_bounds__(256) void knn_kernel(
    const _Float16* __restrict__ xf16, const float* __restrict__ sqg,
    int* __restrict__ idx) {
  __shared__ _Float16 Xs[MP * 64];
  __shared__ float sqs[MP];
  __shared__ float Gs[32 * MP];
  int b = blockIdx.x;
  int tid = threadIdx.x;
  for (int t = tid; t < MP * 64; t += 256) Xs[t] = xf16[b * MP * 64 + t];
  for (int t = tid; t < MP; t += 256) sqs[t] = sqg[b * MP + t];
  __syncthreads();
  int w = tid >> 5;
  int lane = tid & 31;
  for (int cs = 0; cs < 4; ++cs) {
    // 2 row-tiles x 8 col-tiles = 16 tiles; 2 per wave
    for (int tt = w; tt < 16; tt += 8) {
      int rt = tt >> 3, ct = tt & 7;
      v8f acc = {};
      const _Float16* arow = Xs + (cs * 32 + rt * 16) * 64;
      const _Float16* brow = Xs + (ct * 16) * 64;
#pragma unroll
      for (int k0 = 0; k0 < 64; k0 += 32) {
        v16h a = load_frag_a(arow + k0, 64);
        v16h bbv = load_frag_a(brow + k0, 64);
        acc = wmma_f16(a, bbv, acc);
      }
      int col = lane & 15;
      int rh = (lane >> 4) * 8;
#pragma unroll
      for (int r = 0; r < 8; ++r)
        Gs[(rt * 16 + r + rh) * MP + ct * 16 + col] = acc[r];
    }
    __syncthreads();
    if (tid < 32) {
      int m = cs * 32 + tid;
      if (m < MM) {
        float bd[KNN];
        int bi[KNN];
#pragma unroll
        for (int k = 0; k < KNN; ++k) { bd[k] = 1e30f; bi[k] = 0; }
        float sm = sqs[m];
        for (int j = 0; j < MM; ++j) {
          float d = sm + sqs[j] - 2.f * Gs[tid * MP + j];
          if (d < bd[KNN - 1]) {
            int p = KNN - 1;
            while (p > 0 && d < bd[p - 1]) {
              bd[p] = bd[p - 1]; bi[p] = bi[p - 1]; --p;
            }
            bd[p] = d; bi[p] = j;
          }
        }
#pragma unroll
        for (int k = 0; k < KNN; ++k) idx[(b * MP + m) * KNN + k] = bi[k];
      } else {
#pragma unroll
        for (int k = 0; k < KNN; ++k) idx[(b * MP + m) * KNN + k] = 0;
      }
    }
    __syncthreads();
  }
}

// ---------------------------------------------------------------------------
// K3: edge MLP (128->128) on 16 points x 5 neighbors = 80 rows, WMMA,
//     ReLU -> BN2 -> max over K, writes x2 half of h.
// ---------------------------------------------------------------------------
__global__ __launch_bounds__(256) void edge_kernel(
    const float* __restrict__ x1, const int* __restrict__ idx,
    const _Float16* __restrict__ ew16, const float* __restrict__ eb,
    const float* __restrict__ g2, const float* __restrict__ b2,
    const float* __restrict__ m2, const float* __restrict__ v2,
    _Float16* __restrict__ h16) {
  __shared__ _Float16 ef[80 * 128];
  __shared__ float outb[80 * 128];
  int b = blockIdx.x >> 3;
  int mbase = (blockIdx.x & 7) * 16;
  int tid = threadIdx.x;
  // build edge features [xi | xj - xi]
  for (int t = tid; t < 80 * 128; t += 256) {
    int r = t >> 7, col = t & 127;
    int p = r / KNN, k = r % KNN;
    int m = mbase + p;
    int mc = m > (MM - 1) ? (MM - 1) : m;
    int j = idx[(b * MP + mc) * KNN + k];
    float val;
    if (col < 64) {
      val = x1[(b * MP + mc) * 64 + col];
    } else {
      int c2 = col - 64;
      val = x1[(b * MP + j) * 64 + c2] - x1[(b * MP + mc) * 64 + c2];
    }
    ef[t] = (_Float16)val;
  }
  __syncthreads();
  int w = tid >> 5;
  int lane = tid & 31;
  for (int tt = w; tt < 40; tt += 8) {  // 5 row-tiles x 8 col-tiles
    int rt = tt / 8, ct = tt % 8;
    v8f acc = {};
#pragma unroll
    for (int k0 = 0; k0 < 128; k0 += 32) {
      v16h a = load_frag_a(ef + rt * 16 * 128 + k0, 128);
      // packed weights: tn = 8 col tiles
      v16h bf = load_frag_a(ew16 + (size_t)((k0 >> 5) * 8 + ct) * 512, 32);
      acc = wmma_f16(a, bf, acc);
    }
    int col = ct * 16 + (lane & 15);
    float bi = eb[col];
    float sc = rsqrtf(v2[col] + EPS) * g2[col];
    float me = m2[col], be = b2[col];
    int rh = (lane >> 4) * 8;
#pragma unroll
    for (int r = 0; r < 8; ++r) {
      float v = fmaxf(acc[r] + bi, 0.f);          // ReLU first
      v = (v - me) * sc + be;                     // then BN
      outb[(rt * 16 + r + rh) * 128 + col] = v;
    }
  }
  __syncthreads();
  for (int t = tid; t < 16 * 128; t += 256) {
    int p = t >> 7, col = t & 127;
    int m = mbase + p;
    float mx = 0.f;
    if (m < MM) {
      mx = outb[(p * KNN) * 128 + col];
#pragma unroll
      for (int k = 1; k < KNN; ++k)
        mx = fmaxf(mx, outb[(p * KNN + k) * 128 + col]);
    }
    h16[(b * MP + m) * 192 + 64 + col] = (_Float16)mx;
  }
}

// ---------------------------------------------------------------------------
// K4: lin1 (192->1024) + ReLU -> BN3 -> masked max over M -> g (B,1024).
// One workgroup = one batch x 64 output cols. 8 waves: 4 col-tiles x 2 halves.
// ---------------------------------------------------------------------------
__global__ __launch_bounds__(256) void lin1_kernel(
    const _Float16* __restrict__ h16, const _Float16* __restrict__ lw16,
    const float* __restrict__ lb, const float* __restrict__ g3,
    const float* __restrict__ b3, const float* __restrict__ m3,
    const float* __restrict__ v3, float* __restrict__ gout,
    _Float16* __restrict__ g16) {
  __shared__ float red[8][32];
  int b = blockIdx.x >> 4;
  int nb = (blockIdx.x & 15) * 64;
  int tid = threadIdx.x;
  int w = tid >> 5, lane = tid & 31;
  int ct = w & 3, half = w >> 2;
  int gct = (nb >> 4) + ct;        // global column-tile index (tn = 64)
  int colg = gct * 16 + (lane & 15);
  float bi = lb[colg];
  float sc = rsqrtf(v3[colg] + EPS) * g3[colg];
  float me = m3[colg], be = b3[colg];
  int rh = (lane >> 4) * 8;
  float runmax = -1e30f;
  for (int i = 0; i < 4; ++i) {
    int rt = half * 4 + i;
    v8f acc = {};
    const _Float16* abase = h16 + (size_t)(b * MP + rt * 16) * 192;
#pragma unroll
    for (int k0 = 0; k0 < 192; k0 += 32) {
      v16h a = load_frag_a(abase + k0, 192);
      v16h bf = load_frag_a(lw16 + (size_t)((k0 >> 5) * 64 + gct) * 512, 32);
      acc = wmma_f16(a, bf, acc);
    }
#pragma unroll
    for (int r = 0; r < 8; ++r) {
      int mr = rt * 16 + r + rh;
      if (mr < MM) {
        float v = fmaxf(acc[r] + bi, 0.f);   // ReLU
        v = (v - me) * sc + be;              // BN
        runmax = fmaxf(runmax, v);
      }
    }
  }
  red[w][lane] = runmax;
  __syncthreads();
  if (tid < 64) {
    int ctf = tid >> 4, cl = tid & 15;
    float v = fmaxf(fmaxf(red[ctf][cl], red[ctf][cl + 16]),
                    fmaxf(red[ctf + 4][cl], red[ctf + 4][cl + 16]));
    gout[b * 1024 + nb + tid] = v;
    g16[b * 1024 + nb + tid] = (_Float16)v;
  }
}

// ---------------------------------------------------------------------------
// K5/K6: dense GEMM (512 rows) with ReLU -> BN epilogue; one tile per wave.
// Deep K -> prefetch next packed-weight tile (global_prefetch_b8 path).
// ---------------------------------------------------------------------------
__global__ __launch_bounds__(256) void mlp_kernel(
    const _Float16* __restrict__ A, const _Float16* __restrict__ W,
    const float* __restrict__ bias, const float* __restrict__ gg,
    const float* __restrict__ bb, const float* __restrict__ mm,
    const float* __restrict__ vv, _Float16* __restrict__ out16,
    float* __restrict__ out32, int N, int Kd) {
  int w = threadIdx.x >> 5, lane = threadIdx.x & 31;
  int tiles_n = N >> 4;
  int tile = blockIdx.x * 8 + w;
  int rt = tile / tiles_n, ct = tile % tiles_n;
  v8f acc = {};
  for (int k0 = 0; k0 < Kd; k0 += 32) {
    if (k0 + 32 < Kd)
      __builtin_prefetch(W + (size_t)(((k0 + 32) >> 5) * tiles_n + ct) * 512,
                         0, 1);
    v16h a = load_frag_a(A + (size_t)(rt * 16) * Kd + k0, Kd);
    v16h bf = load_frag_a(W + (size_t)((k0 >> 5) * tiles_n + ct) * 512, 32);
    acc = wmma_f16(a, bf, acc);
  }
  int col = ct * 16 + (lane & 15);
  float bi = bias[col];
  float sc = rsqrtf(vv[col] + EPS) * gg[col];
  float me = mm[col], be = bb[col];
  int rh = (lane >> 4) * 8;
#pragma unroll
  for (int r = 0; r < 8; ++r) {
    int mr = rt * 16 + r + rh;
    float v = fmaxf(acc[r] + bi, 0.f);   // ReLU
    v = (v - me) * sc + be;              // BN
    if (out16) out16[(size_t)mr * N + col] = (_Float16)v;
    if (out32) out32[(size_t)mr * N + col] = v;
  }
}

// ---------------------------------------------------------------------------
// K7: head (256 -> 2) on VALU
// ---------------------------------------------------------------------------
__global__ void out_kernel(const float* __restrict__ t2,
                           const float* __restrict__ ow,
                           const float* __restrict__ ob,
                           float* __restrict__ out) {
  int gid = blockIdx.x * 256 + threadIdx.x;
  if (gid >= BB * 2) return;
  int r = gid >> 1, c = gid & 1;
  float s = ob[c];
  const float* row = t2 + r * 256;
#pragma unroll 8
  for (int i = 0; i < 256; ++i) s = fmaf(row[i], ow[i * 2 + c], s);
  out[gid] = s;
}

// ---------------------------------------------------------------------------
// Launch
// ---------------------------------------------------------------------------
extern "C" void kernel_launch(void* const* d_in, const int* in_sizes, int n_in,
                              void* d_out, int out_size, void* d_ws,
                              size_t ws_size, hipStream_t stream) {
  // setup_inputs() order:
  // 0:pos 1:eidx 2:batch 3:lengths 4:conv1_w 5:conv1_b 6:bn1_g 7:bn1_b
  // 8:bn1_m 9:bn1_v 10:edge_w 11:edge_b 12:bn2_g 13:bn2_b 14:bn2_m 15:bn2_v
  // 16:lin1_w 17:lin1_b 18:bn3_g 19:bn3_b 20:bn3_m 21:bn3_v 22:m1_w 23:m1_b
  // 24:bn4_g 25:bn4_b 26:bn4_m 27:bn4_v 28:m2_w 29:m2_b 30:bn5_g 31:bn5_b
  // 32:bn5_m 33:bn5_v 34:out_w 35:out_b
  const float* pos     = (const float*)d_in[0];
  const float* conv1_w = (const float*)d_in[4];
  const float* conv1_b = (const float*)d_in[5];
  const float* bn1_g   = (const float*)d_in[6];
  const float* bn1_b   = (const float*)d_in[7];
  const float* bn1_m   = (const float*)d_in[8];
  const float* bn1_v   = (const float*)d_in[9];
  const float* edge_w  = (const float*)d_in[10];
  const float* edge_b  = (const float*)d_in[11];
  const float* bn2_g   = (const float*)d_in[12];
  const float* bn2_b   = (const float*)d_in[13];
  const float* bn2_m   = (const float*)d_in[14];
  const float* bn2_v   = (const float*)d_in[15];
  const float* lin1_w  = (const float*)d_in[16];
  const float* lin1_b  = (const float*)d_in[17];
  const float* bn3_g   = (const float*)d_in[18];
  const float* bn3_b   = (const float*)d_in[19];
  const float* bn3_m   = (const float*)d_in[20];
  const float* bn3_v   = (const float*)d_in[21];
  const float* m1_w    = (const float*)d_in[22];
  const float* m1_b    = (const float*)d_in[23];
  const float* bn4_g   = (const float*)d_in[24];
  const float* bn4_b   = (const float*)d_in[25];
  const float* bn4_m   = (const float*)d_in[26];
  const float* bn4_v   = (const float*)d_in[27];
  const float* m2_w    = (const float*)d_in[28];
  const float* m2_b    = (const float*)d_in[29];
  const float* bn5_g   = (const float*)d_in[30];
  const float* bn5_b   = (const float*)d_in[31];
  const float* bn5_m   = (const float*)d_in[32];
  const float* bn5_v   = (const float*)d_in[33];
  const float* out_w   = (const float*)d_in[34];
  const float* out_b   = (const float*)d_in[35];

  char* ws = (char*)d_ws;
  _Float16* ew16  = (_Float16*)(ws + 0);          //  32768 B (packed tiles)
  _Float16* lw16  = (_Float16*)(ws + 32768);      // 393216 B (packed tiles)
  _Float16* m1w16 = (_Float16*)(ws + 425984);     // 1048576 B (packed tiles)
  _Float16* m2w16 = (_Float16*)(ws + 1474560);    // 262144 B (packed tiles)
  float*    x1f   = (float*)   (ws + 1736704);    // 16777216 B
  _Float16* xf16  = (_Float16*)(ws + 18513920);   // 8388608 B
  float*    sqg   = (float*)   (ws + 26902528);   // 262144 B
  int*      idx   = (int*)     (ws + 27164672);   // 1310720 B
  _Float16* h16   = (_Float16*)(ws + 28475392);   // 25165824 B
  float*    gf32  = (float*)   (ws + 53641216);   // 2097152 B
  _Float16* g16   = (_Float16*)(ws + 55738368);   // 1048576 B
  _Float16* t1f16 = (_Float16*)(ws + 56786944);   // 524288 B
  float*    t2f32 = (float*)   (ws + 57311232);   // 524288 B
  (void)ws_size; (void)in_sizes; (void)n_in; (void)out_size;

  pack_b_kernel<<<(16384 + 255) / 256, 256, 0, stream>>>(edge_w, ew16, 128, 128);
  pack_b_kernel<<<(196608 + 255) / 256, 256, 0, stream>>>(lin1_w, lw16, 192, 1024);
  pack_b_kernel<<<(524288 + 255) / 256, 256, 0, stream>>>(m1_w, m1w16, 1024, 512);
  pack_b_kernel<<<(131072 + 255) / 256, 256, 0, stream>>>(m2_w, m2w16, 512, 256);

  conv_fuse_kernel<<<(BB * MP * 64) / 256, 256, 0, stream>>>(
      pos, conv1_w, conv1_b, bn1_g, bn1_b, bn1_m, bn1_v, x1f, xf16, h16);

  sq_kernel<<<(BB * MP) / 256, 256, 0, stream>>>(x1f, sqg);

  knn_kernel<<<BB, 256, 0, stream>>>(xf16, sqg, idx);

  edge_kernel<<<BB * 8, 256, 0, stream>>>(x1f, idx, ew16, edge_b, bn2_g, bn2_b,
                                          bn2_m, bn2_v, h16);

  lin1_kernel<<<BB * 16, 256, 0, stream>>>(h16, lw16, lin1_b, bn3_g, bn3_b,
                                           bn3_m, bn3_v, gf32, g16);

  // m1: (512x1024)@(1024x512) -> 32x32 tiles = 1024 tiles / 8 waves
  mlp_kernel<<<128, 256, 0, stream>>>(g16, m1w16, m1_b, bn4_g, bn4_b, bn4_m,
                                      bn4_v, t1f16, nullptr, 512, 1024);
  // m2: (512x512)@(512x256) -> 32x16 tiles = 512 tiles / 8 waves
  mlp_kernel<<<64, 256, 0, stream>>>(t1f16, m2w16, m2_b, bn5_g, bn5_b, bn5_m,
                                     bn5_v, nullptr, t2f32, 256, 512);

  out_kernel<<<4, 256, 0, stream>>>(t2f32, out_w, out_b, (float*)d_out);
}
